// GINEConvEncoder_61615600828639
// MI455X (gfx1250) — matrix operand order repaired and checked by reading, not compiled, not run on hardware
//
#include <hip/hip_runtime.h>
#include <math.h>

// ---------------------------------------------------------------------------
// GINEConv encoder for MI455X (gfx1250, wave32).
// All GEMMs run through v_wmma_f32_16x16x32_bf16 (bf16 in, f32 accumulate):
// with bf16 WMMA the 36 GFLOP of GEMM work is far below the ~80us of HBM
// traffic (23.3 TB/s), so the kernel is memory-bound -- the right operating
// point for this op on CDNA5.
// ---------------------------------------------------------------------------

typedef __bf16 bf16;
typedef __attribute__((ext_vector_type(16))) __bf16 v16bf;
typedef __attribute__((ext_vector_type(8)))  __bf16 v8bf;
typedef __attribute__((ext_vector_type(8)))  float  v8f;

#define WMMA_BF16(A, B, C) \
  __builtin_amdgcn_wmma_f32_16x16x32_bf16(false, (A), false, (B), (short)0, (C), false, false)

constexpr int Dh   = 128;       // hidden
constexpr int EDh  = 64;        // edge dim
constexpr int Gn   = 128;       // graphs
constexpr int N0   = 512;       // nodes/graph at layer 0
constexpr int EGn  = 4096;      // edges/graph
constexpr int Etot = Gn * EGn;  // 524288

// ---------------------------------------------------------------------------
// small utility kernels
// ---------------------------------------------------------------------------
__global__ void k_zero_f(float* p, int n) {
  int i = blockIdx.x * blockDim.x + threadIdx.x;
  if (i < n) p[i] = 0.0f;
}
__global__ void k_fill_i(int* p, int v, int n) {
  int i = blockIdx.x * blockDim.x + threadIdx.x;
  if (i < n) p[i] = v;
}
__global__ void k_init_edges(const int* __restrict__ es, const int* __restrict__ ed,
                             int* __restrict__ src, int* __restrict__ dst,
                             int* __restrict__ valid, int n) {
  int i = blockIdx.x * blockDim.x + threadIdx.x;
  if (i < n) { src[i] = es[i]; dst[i] = ed[i]; valid[i] = 1; }
}

// ---------------------------------------------------------------------------
// Edge stage: e = edge_attr @ We + be ; msg = relu(x[src] + e) ; agg[dst] += msg
// One wave per 16-edge tile: A = 16x64 bf16 (from f32 edge_attr),
// B = 64x128 weights swizzled in LDS, 2 WMMAs per 16-col output tile.
// ---------------------------------------------------------------------------
__global__ __launch_bounds__(256)
void k_edge(const float* __restrict__ edge_attr,  // [E,64] f32
            const float* __restrict__ We,         // [64,128] f32
            const float* __restrict__ be,         // [128]
            const float* __restrict__ x,          // [Ncur,128]
            const int*   __restrict__ src,
            const int*   __restrict__ dst,
            const int*   __restrict__ valid,
            float*       __restrict__ agg)        // [Ncur,128]
{
  // W swizzled so each lane's B fragment is one contiguous 32B chunk:
  // sW[((kt*8+nt)*32 + lane)*16 + j] = W[kt*32 + (lane>>4)*8 + (j&7) + (j>>3)*16][nt*16 + (lane&15)]
  __shared__ __attribute__((aligned(32))) bf16 sW[EDh * Dh];   // 16 KB
  __shared__ float sB[Dh];

  int tid = threadIdx.x;
  for (int i = tid; i < EDh * Dh; i += 256) {
    int j  = i & 15;
    int ln = (i >> 4) & 31;
    int fr = i >> 9;              // 0..15
    int kt = fr >> 3, nt = fr & 7;
    int kk = kt * 32 + (ln >> 4) * 8 + (j & 7) + (j >> 3) * 16;
    int nn = nt * 16 + (ln & 15);
    sW[i] = (bf16)We[kk * Dh + nn];
  }
  for (int i = tid; i < Dh; i += 256) sB[i] = be[i];
  __syncthreads();

  int wave = tid >> 5;
  int lane = tid & 31;
  int hf   = lane >> 4;      // lane half
  int l16  = lane & 15;
  int e0   = (blockIdx.x * 8 + wave) * 16;   // Etot % 128 == 0 -> always full

  // A fragments: row M = l16; low lanes hold K 0..7 & 16..23 of each 32-chunk
  v16bf a[2];
  {
    const float* ea = edge_attr + (size_t)(e0 + l16) * EDh;
#pragma unroll
    for (int kt = 0; kt < 2; ++kt) {
      int kb = kt * 32 + hf * 8;
#pragma unroll
      for (int j = 0; j < 8; ++j) a[kt][j]     = (bf16)ea[kb + j];
#pragma unroll
      for (int j = 0; j < 8; ++j) a[kt][8 + j] = (bf16)ea[kb + 16 + j];
    }
  }

  // per-lane edge metadata for the 8 C rows (M = r + 8*hf)
  int m_src[8], m_dst[8], m_val[8];
#pragma unroll
  for (int r = 0; r < 8; ++r) {
    int e = e0 + r + 8 * hf;
    m_src[r] = src[e]; m_dst[r] = dst[e]; m_val[r] = valid[e];
  }

  for (int nt = 0; nt < 8; ++nt) {
    int n = nt * 16 + l16;
    v8f c;
    float bb = sB[n];
#pragma unroll
    for (int r = 0; r < 8; ++r) c[r] = bb;
    v16bf b0 = *(const v16bf*)(sW + ((0 * 8 + nt) * 32 + lane) * 16);
    v16bf b1 = *(const v16bf*)(sW + ((1 * 8 + nt) * 32 + lane) * 16);
    c = WMMA_BF16(a[0], b0, c);
    c = WMMA_BF16(a[1], b1, c);
#pragma unroll
    for (int r = 0; r < 8; ++r) {
      if (m_val[r]) {
        float msg = c[r] + x[(size_t)m_src[r] * Dh + n];
        msg = fmaxf(msg, 0.0f);
        atomicAdd(&agg[(size_t)m_dst[r] * Dh + n], msg);
      }
    }
  }
}

// ---------------------------------------------------------------------------
// MLP stage: h = relu((x+agg)@W1+b1)@W2+b2, written in-place over agg.
// 4 waves / 64 rows per block; W1 then W2 swizzled in LDS; the intermediate
// bf16 tile is staged in LDS to re-shape C-layout -> A-layout between GEMMs.
// Also accumulates per-channel BN sum / sumsq.
// ---------------------------------------------------------------------------
__global__ __launch_bounds__(128)
void k_mlp(const float* __restrict__ x,       // [Ncur,128]
           float*       __restrict__ aggh,    // in: agg, out: h  [Ncur,128]
           const float* __restrict__ W1, const float* __restrict__ b1,
           const float* __restrict__ W2, const float* __restrict__ b2,
           float* __restrict__ bn_sum, float* __restrict__ bn_sq)
{
  __shared__ __attribute__((aligned(32))) bf16 sW[Dh * Dh];        // 32 KB (W1 then W2)
  __shared__ __attribute__((aligned(32))) bf16 stage[4][16 * Dh];  // 16 KB
  __shared__ float sS[Dh], sQ[Dh];

  int tid  = threadIdx.x;
  int wave = tid >> 5;
  int lane = tid & 31;
  int hf   = lane >> 4;
  int l16  = lane & 15;
  int m0   = (blockIdx.x * 4 + wave) * 16;

  if (tid < Dh) { sS[tid] = 0.0f; sQ[tid] = 0.0f; }

  // stage W1 (swizzled, f32 -> bf16)
  for (int i = tid; i < Dh * Dh; i += 128) {
    int j  = i & 15;
    int ln = (i >> 4) & 31;
    int fr = i >> 9;              // 0..31
    int kt = fr >> 3, nt = fr & 7;
    sW[i] = (bf16)W1[(kt * 32 + (ln >> 4) * 8 + (j & 7) + (j >> 3) * 16) * Dh + nt * 16 + (ln & 15)];
  }
  // stage h_in = x + agg as bf16 (row-major 16x128 per wave)
  {
    const float* xr = x    + (size_t)m0 * Dh;
    const float* ar = aggh + (size_t)m0 * Dh;
    for (int i = lane; i < 16 * Dh; i += 32)
      stage[wave][i] = (bf16)(xr[i] + ar[i]);
  }
  __syncthreads();

  // A fragments from the staged row-major tile
  v16bf a[4];
#pragma unroll
  for (int kt = 0; kt < 4; ++kt) {
    const bf16* rb = &stage[wave][l16 * Dh + kt * 32 + hf * 8];
    v8bf lo = *(const v8bf*)(rb);
    v8bf hi = *(const v8bf*)(rb + 16);
    a[kt] = __builtin_shufflevector(lo, hi, 0,1,2,3,4,5,6,7,8,9,10,11,12,13,14,15);
  }

  // GEMM1 + ReLU -> stage as bf16
  for (int nt = 0; nt < 8; ++nt) {
    int n = nt * 16 + l16;
    v8f c;
    float bb = b1[n];
#pragma unroll
    for (int r = 0; r < 8; ++r) c[r] = bb;
#pragma unroll
    for (int kt = 0; kt < 4; ++kt) {
      v16bf b = *(const v16bf*)(sW + ((kt * 8 + nt) * 32 + lane) * 16);
      c = WMMA_BF16(a[kt], b, c);
    }
#pragma unroll
    for (int r = 0; r < 8; ++r)
      stage[wave][(r + 8 * hf) * Dh + n] = (bf16)fmaxf(c[r], 0.0f);
  }
  __syncthreads();

  // overwrite LDS with swizzled W2
  for (int i = tid; i < Dh * Dh; i += 128) {
    int j  = i & 15;
    int ln = (i >> 4) & 31;
    int fr = i >> 9;
    int kt = fr >> 3, nt = fr & 7;
    sW[i] = (bf16)W2[(kt * 32 + (ln >> 4) * 8 + (j & 7) + (j >> 3) * 16) * Dh + nt * 16 + (ln & 15)];
  }
  __syncthreads();

#pragma unroll
  for (int kt = 0; kt < 4; ++kt) {
    const bf16* rb = &stage[wave][l16 * Dh + kt * 32 + hf * 8];
    v8bf lo = *(const v8bf*)(rb);
    v8bf hi = *(const v8bf*)(rb + 16);
    a[kt] = __builtin_shufflevector(lo, hi, 0,1,2,3,4,5,6,7,8,9,10,11,12,13,14,15);
  }

  // GEMM2 + store + BN partials
  for (int nt = 0; nt < 8; ++nt) {
    int n = nt * 16 + l16;
    v8f c;
    float bb = b2[n];
#pragma unroll
    for (int r = 0; r < 8; ++r) c[r] = bb;
#pragma unroll
    for (int kt = 0; kt < 4; ++kt) {
      v16bf b = *(const v16bf*)(sW + ((kt * 8 + nt) * 32 + lane) * 16);
      c = WMMA_BF16(a[kt], b, c);
    }
    float ls = 0.0f, lq = 0.0f;
#pragma unroll
    for (int r = 0; r < 8; ++r) {
      float v = c[r];
      aggh[(size_t)(m0 + r + 8 * hf) * Dh + n] = v;
      ls += v; lq += v * v;
    }
    atomicAdd(&sS[n], ls);
    atomicAdd(&sQ[n], lq);
  }
  __syncthreads();
  if (tid < Dh) { atomicAdd(&bn_sum[tid], sS[tid]); atomicAdd(&bn_sq[tid], sQ[tid]); }
}

// ---------------------------------------------------------------------------
// BN statistics finalize + pool-weight norm
// bn_stats layout: [0..127]=sum [128..255]=sumsq [256..383]=mu [384..511]=rsqrt
// ---------------------------------------------------------------------------
__global__ __launch_bounds__(128)
void k_bn_stats(float* __restrict__ bn, float invN,
                const float* __restrict__ pool_w, float* __restrict__ pw_inv)
{
  __shared__ float red[128];
  int t = threadIdx.x;
  float m = bn[t] * invN;
  float v = bn[128 + t] * invN - m * m;
  bn[256 + t] = m;
  bn[384 + t] = rsqrtf(v + 1e-5f);
  float pw = pool_w[t];
  red[t] = pw * pw;
  __syncthreads();
  for (int s = 64; s > 0; s >>= 1) { if (t < s) red[t] += red[t + s]; __syncthreads(); }
  if (t == 0) *pw_inv = rsqrtf(red[0]);
}

// ---------------------------------------------------------------------------
// BN apply + ReLU (in place) + per-row pooling score = tanh(h.w/||w||)
// one wave per row
// ---------------------------------------------------------------------------
__global__ __launch_bounds__(256)
void k_bn_relu_score(float* __restrict__ h, const float* __restrict__ bn,
                     const float* __restrict__ g, const float* __restrict__ b,
                     const float* __restrict__ pool_w, const float* __restrict__ pw_inv,
                     float* __restrict__ score)
{
  int wave = threadIdx.x >> 5, lane = threadIdx.x & 31;
  int row  = blockIdx.x * 8 + wave;
  float* hr = h + (size_t)row * Dh;
  float dot = 0.0f;
#pragma unroll
  for (int i = 0; i < 4; ++i) {
    int c = lane + 32 * i;
    float v = (hr[c] - bn[256 + c]) * bn[384 + c] * g[c] + b[c];
    v = fmaxf(v, 0.0f);
    hr[c] = v;
    dot += v * pool_w[c];
  }
#pragma unroll
  for (int o = 16; o > 0; o >>= 1) dot += __shfl_xor(dot, o, 32);
  if (lane == 0) score[row] = tanhf(dot * (*pw_inv));
}

// ---------------------------------------------------------------------------
// Per-graph: bitonic top-k selection, gated gather into x_next, new-index map,
// attention gate + softmax, readout accumulation into out.
// One 512-thread block per graph. (Output is invariant to kept-node ORDER, so
// any consistent top-k selection matches jax.lax.top_k for this pipeline.)
// ---------------------------------------------------------------------------
__global__ __launch_bounds__(512)
void k_topk(const float* __restrict__ h, const float* __restrict__ score,
            float* __restrict__ x_next, int* __restrict__ new_idx,
            const float* __restrict__ gate_w, const float* __restrict__ gate_b_p,
            float* __restrict__ out, int n, int k)
{
  __shared__ float s[512];
  __shared__ int   id[512];
  __shared__ float gt[512];
  __shared__ float red[512];
  int g = blockIdx.x;
  int t = threadIdx.x;

  s[t]  = (t < n) ? score[g * n + t] : -3.4e38f;
  id[t] = t;
  // bitonic sort, descending
  for (int size = 2; size <= 512; size <<= 1) {
    for (int stride = size >> 1; stride > 0; stride >>= 1) {
      __syncthreads();
      int i = t, j = i ^ stride;
      if (j > i) {
        bool up = ((i & size) == 0);
        float si = s[i], sj = s[j];
        if ((si < sj) == up) {
          s[i] = sj; s[j] = si;
          int tmp = id[i]; id[i] = id[j]; id[j] = tmp;
        }
      }
    }
  }
  __syncthreads();

  if (t < k) new_idx[g * n + id[t]] = g * k + t;
  // gated gather: x_next[g*k + j] = h[g*n + id[j]] * s[j]
  for (int idx = t; idx < k * Dh; idx += 512) {
    int j = idx >> 7, c = idx & 127;
    x_next[(size_t)(g * k + j) * Dh + c] =
        h[(size_t)(g * n + id[j]) * Dh + c] * s[j];
  }
  __threadfence_block();
  __syncthreads();

  // attention gate
  float gb = *gate_b_p;
  if (t < k) {
    const float* xr = x_next + (size_t)(g * k + t) * Dh;
    float d = 0.0f;
    for (int c = 0; c < Dh; ++c) d += xr[c] * gate_w[c];
    gt[t] = d + gb;
  } else {
    gt[t] = -3.4e38f;
  }
  __syncthreads();
  // softmax over k
  red[t] = gt[t];
  __syncthreads();
  for (int o = 256; o > 0; o >>= 1) { if (t < o) red[t] = fmaxf(red[t], red[t + o]); __syncthreads(); }
  float mx = red[0];
  __syncthreads();
  float ex = (t < k) ? __expf(gt[t] - mx) : 0.0f;
  gt[t]  = ex;
  red[t] = ex;
  __syncthreads();
  for (int o = 256; o > 0; o >>= 1) { if (t < o) red[t] += red[t + o]; __syncthreads(); }
  float inv = 1.0f / red[0];
  __syncthreads();
  // readout: out[g] += sum_j attn[j] * x_next[g,j,:]
  if (t < Dh) {
    float acc = 0.0f;
    for (int j = 0; j < k; ++j)
      acc += gt[j] * x_next[(size_t)(g * k + j) * Dh + t];
    out[g * Dh + t] += acc * inv;
  }
}

// ---------------------------------------------------------------------------
// Edge remap to pooled numbering; invalidate edges touching removed nodes.
// ---------------------------------------------------------------------------
__global__ void k_remap(int* __restrict__ src, int* __restrict__ dst,
                        int* __restrict__ valid, const int* __restrict__ new_idx, int n)
{
  int i = blockIdx.x * blockDim.x + threadIdx.x;
  if (i < n && valid[i]) {
    int s2 = new_idx[src[i]], d2 = new_idx[dst[i]];
    if (s2 >= 0 && d2 >= 0) { src[i] = s2; dst[i] = d2; }
    else { valid[i] = 0; src[i] = 0; dst[i] = 0; }
  }
}

// ---------------------------------------------------------------------------
extern "C" void kernel_launch(void* const* d_in, const int* in_sizes, int n_in,
                              void* d_out, int out_size, void* d_ws, size_t ws_size,
                              hipStream_t stream)
{
  const float* x_in    = (const float*)d_in[0];
  const float* eattr   = (const float*)d_in[1];
  const float* lin_e_w = (const float*)d_in[2];
  const float* lin_e_b = (const float*)d_in[3];
  const float* mlp_w1  = (const float*)d_in[4];
  const float* mlp_b1  = (const float*)d_in[5];
  const float* mlp_w2  = (const float*)d_in[6];
  const float* mlp_b2  = (const float*)d_in[7];
  const float* bn_g    = (const float*)d_in[8];
  const float* bn_b    = (const float*)d_in[9];
  const float* pool_w  = (const float*)d_in[10];
  const float* gate_w  = (const float*)d_in[11];
  const float* gate_b  = (const float*)d_in[12];
  const int*   e_src   = (const int*)d_in[13];
  const int*   e_dst   = (const int*)d_in[14];
  float* out = (float*)d_out;

  char* w = (char*)d_ws;
  auto carve = [&](size_t bytes) -> char* {
    char* p = w; w += (bytes + 255) & ~(size_t)255; return p;
  };
  const size_t xbytes = (size_t)Gn * N0 * Dh * sizeof(float);   // 33.5 MB
  float* xa     = (float*)carve(xbytes);
  float* xb     = (float*)carve(xbytes);
  float* aggh   = (float*)carve(xbytes);
  int*   srcb   = (int*)carve((size_t)Etot * 4);
  int*   dstb   = (int*)carve((size_t)Etot * 4);
  int*   validb = (int*)carve((size_t)Etot * 4);
  float* score  = (float*)carve((size_t)Gn * N0 * 4);
  int*   nidx   = (int*)carve((size_t)Gn * N0 * 4);
  float* bn     = (float*)carve(512 * sizeof(float));   // sum | sumsq | mu | rs
  float* pw_inv = (float*)carve(256);

  // init (deterministic every call)
  hipMemcpyAsync(xa, x_in, xbytes, hipMemcpyDeviceToDevice, stream);
  k_init_edges<<<Etot / 256, 256, 0, stream>>>(e_src, e_dst, srcb, dstb, validb, Etot);
  k_zero_f<<<(Gn * Dh + 255) / 256, 256, 0, stream>>>(out, Gn * Dh);

  const int ns[4] = {512, 410, 328, 263};   // ceil(0.8*n) chain
  float* xcur = xa;
  float* xnxt = xb;
  for (int l = 0; l < 3; ++l) {
    int n = ns[l], k = ns[l + 1];
    int Ncur = Gn * n;

    k_zero_f<<<(Ncur * Dh + 255) / 256, 256, 0, stream>>>(aggh, Ncur * Dh);
    k_zero_f<<<1, 256, 0, stream>>>(bn, 256);
    k_fill_i<<<(Ncur + 255) / 256, 256, 0, stream>>>(nidx, -1, Ncur);

    k_edge<<<Etot / 128, 256, 0, stream>>>(
        eattr, lin_e_w + (size_t)l * EDh * Dh, lin_e_b + (size_t)l * Dh,
        xcur, srcb, dstb, validb, aggh);

    k_mlp<<<Ncur / 64, 128, 0, stream>>>(
        xcur, aggh,
        mlp_w1 + (size_t)l * Dh * Dh, mlp_b1 + (size_t)l * Dh,
        mlp_w2 + (size_t)l * Dh * Dh, mlp_b2 + (size_t)l * Dh,
        bn, bn + 128);

    k_bn_stats<<<1, 128, 0, stream>>>(bn, 1.0f / (float)Ncur, pool_w + (size_t)l * Dh, pw_inv);

    k_bn_relu_score<<<Ncur / 8, 256, 0, stream>>>(
        aggh, bn, bn_g + (size_t)l * Dh, bn_b + (size_t)l * Dh,
        pool_w + (size_t)l * Dh, pw_inv, score);

    k_topk<<<Gn, 512, 0, stream>>>(
        aggh, score, xnxt, nidx,
        gate_w + (size_t)l * Dh, gate_b + l, out, n, k);

    if (l < 2)
      k_remap<<<Etot / 256, 256, 0, stream>>>(srcb, dstb, validb, nidx, Etot);

    float* tmp = xcur; xcur = xnxt; xnxt = tmp;
  }
  (void)in_sizes; (void)n_in; (void)out_size; (void)ws_size;
}
